// PerformerAttention_80874234183848
// MI455X (gfx1250) — matrix operand order repaired and checked by reading, not compiled, not run on hardware
//
#include <hip/hip_runtime.h>
#include <hip/hip_bf16.h>
#include <stdint.h>

typedef __bf16 bf16;
typedef __attribute__((ext_vector_type(16))) __bf16 v16bf;
typedef __attribute__((ext_vector_type(8)))  float  v8f;
typedef __attribute__((ext_vector_type(4)))  unsigned int u32x4;
typedef __attribute__((ext_vector_type(8)))  int i32x8;
typedef __attribute__((ext_vector_type(4)))  int i32x4;

#define NTHREADS 128   // 4 wave32 per workgroup
#define LDT  34        // 32-wide K tile + 1 DWORD pad (matches TDM pad_amount=1)
#define LDT2 66        // 64-wide K tile + 2 pad

#if defined(__AMDGCN__) && __has_builtin(__builtin_amdgcn_tensor_load_to_lds)
#define USE_TDM 1
#else
#define USE_TDM 0
#endif

__device__ __forceinline__ v8f wmma_bf16f32(v16bf a, v16bf b, v8f c) {
  // D = A(16x32 bf16) * B(32x16 bf16) + C(16x16 f32)
  return __builtin_amdgcn_wmma_f32_16x16x32_bf16(false, a, false, b, (short)0, c,
                                                 false, false);
}

// A fragment: rows [m0,m0+16) x K=32 from LDS tile (row stride lda elements).
__device__ __forceinline__ v16bf load_frag_a(const bf16* tile, int lda, int m0, int lane) {
  union { v16bf v; uint32_t u[8]; } r;
  const int m = m0 + (lane & 15);
  const int h = lane >> 4;
  const bf16* row = tile + m * lda;
#pragma unroll
  for (int i = 0; i < 8; ++i) {
    const int k = (i < 4) ? (8 * h + 2 * i) : (16 + 8 * h + 2 * (i - 4));
    __builtin_memcpy(&r.u[i], row + k, 4);
  }
  return r.v;
}

// B fragment: K=32 x cols [n0,n0+16); tile stored transposed as [n][k] (stride ldb).
__device__ __forceinline__ v16bf load_frag_b(const bf16* tile, int ldb, int n0, int lane) {
  union { v16bf v; uint32_t u[8]; } r;
  const int n = n0 + (lane & 15);
  const int h = lane >> 4;
  const bf16* row = tile + n * ldb;
#pragma unroll
  for (int i = 0; i < 8; ++i) {
    const int k = (i < 4) ? (8 * h + 2 * i) : (16 + 8 * h + 2 * (i - 4));
    __builtin_memcpy(&r.u[i], row + k, 4);
  }
  return r.v;
}

#if USE_TDM
// TDM: load a [rows x 32] bf16 tile (row stride 512 elements in a 32768x512 tensor)
// into LDS with 1 DWORD of padding after every 16 DWORDs (i.e. LDS stride = 34 bf16).
__device__ __forceinline__ void tdm_load_tile(const bf16* gsrc, uint32_t lds_byte_addr,
                                              int rows) {
  const unsigned long long ga = (unsigned long long)(uintptr_t)gsrc;
  u32x4 g0;
  g0[0] = 1u;                                   // count=1, user mode
  g0[1] = lds_byte_addr;                        // D#.lds_addr
  g0[2] = (unsigned)(ga & 0xffffffffu);         // D#.global_addr[31:0]
  g0[3] = (unsigned)((ga >> 32) & 0x1ffffffu)   // D#.global_addr[56:32]
        | (2u << 30);                           // type = 2 (image)
  i32x8 g1;
  g1[0] = (1 << 16)        // data_size = 1 -> 2 bytes
        | (1 << 20)        // pad_enable
        | (3 << 22);       // pad_interval = 3 -> every 16 DWORDs (one 64B row)
                           // pad_amount = 0 -> 1 DWORD (the 2-element LDT pad)
  g1[1] = (int)(512u << 16);    // tensor_dim0 = 512 (contiguous row length)
  g1[2] = (int)(32768u << 16);  // tensor_dim1 = 32768 rows (bits[15:0] of dim1)
  g1[3] = (int)(32u << 16);     // dim1 hi = 0 | tile_dim0 = 32
  g1[4] = rows;                 // tile_dim1 = rows, tile_dim2 = 0
  g1[5] = 512;                  // tensor_dim0_stride = 512 elements
  g1[6] = 0;                    // stride hi / tensor_dim1_stride lo
  g1[7] = 0;
  const i32x4 z4 = {0, 0, 0, 0};              // groups 2/3 unused (2-D tile)
  const i32x8 z8 = {0, 0, 0, 0, 0, 0, 0, 0};  // clang-23 6-arg form: extra group
  __builtin_amdgcn_tensor_load_to_lds(g0, g1, z4, z4, z8, 0);
}
#endif

// ---------------- elementwise helpers ----------------
__global__ void k_cvt(const float* __restrict__ s, bf16* __restrict__ d, int n) {
  int i = blockIdx.x * blockDim.x + threadIdx.x;
  if (i < n) d[i] = (bf16)s[i];
}
__global__ void k_zero(float* __restrict__ p, int n) {
  int i = blockIdx.x * blockDim.x + threadIdx.x;
  if (i < n) p[i] = 0.f;
}

// ---------------- stage 1: qkv = x @ w_qkv  ([32768,512]x[512,1536] -> bf16) -----
// 128x64 block tile, each wave computes 32x64 (2 A-frags, 8 accumulators).
__global__ __launch_bounds__(NTHREADS) void k_gemm_qkv(const bf16* __restrict__ A,
                                                       const bf16* __restrict__ W,
                                                       bf16* __restrict__ C) {
  __shared__ __align__(16) bf16 As[128 * LDT];
  __shared__ __align__(16) bf16 Bs[64 * LDT];
  const int tid = threadIdx.x, lane = tid & 31, wave = tid >> 5;
  const long m0 = (long)blockIdx.x * 128;
  const int  n0 = blockIdx.y * 64;
  v8f acc[2][4] = {};
  for (int k0 = 0; k0 < 512; k0 += 32) {
#if USE_TDM
    if (wave == 0) tdm_load_tile(A + m0 * 512 + k0, (uint32_t)(uintptr_t)As, 128);
#else
    for (int idx = tid; idx < 128 * 16; idx += NTHREADS) {
      const int r = idx >> 4, kp = (idx & 15) << 1;
      uint32_t v; __builtin_memcpy(&v, A + (m0 + r) * 512 + k0 + kp, 4);
      __builtin_memcpy(&As[r * LDT + kp], &v, 4);
    }
#endif
    for (int idx = tid; idx < 32 * 32; idx += NTHREADS) {   // B^T: Bs[n][k], paired n
      const int k = idx >> 5, n2 = (idx & 31) << 1;
      uint32_t v; __builtin_memcpy(&v, W + (long)(k0 + k) * 1536 + n0 + n2, 4);
      bf16 two[2]; __builtin_memcpy(two, &v, 4);
      Bs[n2 * LDT + k] = two[0];
      Bs[(n2 + 1) * LDT + k] = two[1];
    }
#if USE_TDM
    if (wave == 0) __builtin_amdgcn_s_wait_tensorcnt(0);
#endif
    __syncthreads();
    const v16bf a0 = load_frag_a(As, LDT, wave * 32, lane);
    const v16bf a1 = load_frag_a(As, LDT, wave * 32 + 16, lane);
#pragma unroll
    for (int g = 0; g < 4; ++g) {
      const v16bf bv = load_frag_b(Bs, LDT, g * 16, lane);
      acc[0][g] = wmma_bf16f32(a0, bv, acc[0][g]);
      acc[1][g] = wmma_bf16f32(a1, bv, acc[1][g]);
    }
    __syncthreads();
  }
  const int hh = lane >> 4, nc = lane & 15;
#pragma unroll
  for (int s = 0; s < 2; ++s) {
#pragma unroll
    for (int g = 0; g < 4; ++g) {
#pragma unroll
      for (int i = 0; i < 8; ++i) {
        const long row = m0 + wave * 32 + s * 16 + i + 8 * hh;
        C[row * 1536 + n0 + g * 16 + nc] = (bf16)acc[s][g][i];
      }
    }
  }
}

// ---------------- stage 2: phi = elu(q|k @ proj[h]) + 1 -------------------------
__global__ __launch_bounds__(NTHREADS) void k_phi(const bf16* __restrict__ QKV,
                                                  const bf16* __restrict__ P,
                                                  bf16* __restrict__ qphi,
                                                  bf16* __restrict__ kphi) {
  __shared__ __align__(16) bf16 As[64 * LDT2];
  __shared__ __align__(16) bf16 Bs[64 * LDT2];
  const int tid = threadIdx.x, lane = tid & 31, wave = tid >> 5;
  const long m0 = (long)blockIdx.x * 64;
  const int h = blockIdx.y, which = blockIdx.z;
  const bf16* src = QKV + m0 * 1536 + which * 512 + h * 64;
  for (int idx = tid; idx < 64 * 32; idx += NTHREADS) {     // A: 64 rows x 64 d
    const int r = idx >> 5, kp = (idx & 31) << 1;
    uint32_t v; __builtin_memcpy(&v, src + (long)r * 1536 + kp, 4);
    __builtin_memcpy(&As[r * LDT2 + kp], &v, 4);
  }
  for (int idx = tid; idx < 64 * 64; idx += NTHREADS) {     // B^T: Bs[f][d] = proj[h,d,f]
    const int d = idx >> 6, f = idx & 63;
    Bs[f * LDT2 + d] = P[(h * 64 + d) * 64 + f];
  }
  __syncthreads();
  v8f acc[4] = {};
#pragma unroll
  for (int ks = 0; ks < 64; ks += 32) {
    const v16bf af = load_frag_a(As + ks, LDT2, wave * 16, lane);
#pragma unroll
    for (int g = 0; g < 4; ++g) {
      const v16bf bv = load_frag_b(Bs + ks, LDT2, g * 16, lane);
      acc[g] = wmma_bf16f32(af, bv, acc[g]);
    }
  }
  bf16* dst = which ? kphi : qphi;
  const int hh = lane >> 4, nc = lane & 15;
#pragma unroll
  for (int g = 0; g < 4; ++g) {
#pragma unroll
    for (int i = 0; i < 8; ++i) {
      const float x = acc[g][i];
      const float ph = (x > 0.f) ? (x + 1.f) : __expf(x);   // elu(x)+1
      const long row = m0 + wave * 16 + i + 8 * hh;
      dst[row * 512 + h * 64 + g * 16 + nc] = (bf16)ph;
    }
  }
}

// ---------------- stage 3: kv[b,h,f,d] = sum_n kphi*v ; ksum[b,h,f] -------------
__global__ __launch_bounds__(NTHREADS) void k_kv(const bf16* __restrict__ kphi,
                                                 const bf16* __restrict__ QKV,
                                                 float* __restrict__ kvbuf,
                                                 float* __restrict__ ksum) {
  __shared__ __align__(16) bf16 As[64 * LDT];  // [f][n_local]
  __shared__ __align__(16) bf16 Bs[64 * LDT];  // [d][n_local]
  const int tid = threadIdx.x, lane = tid & 31, wave = tid >> 5;
  const int bh = blockIdx.x, b = bh >> 3, h = bh & 7;
  const long nbase = (long)b * 4096 + (long)blockIdx.y * 512;  // split-K chunk
  v8f acc[4] = {};
  float ksacc = 0.f;
  for (int kk = 0; kk < 512; kk += 32) {
    const long rowb = nbase + kk;
    for (int idx = tid; idx < 64 * 32; idx += NTHREADS) {
      const int k = idx >> 6, f = idx & 63;
      As[f * LDT + k] = kphi[(rowb + k) * 512 + h * 64 + f];
    }
    for (int idx = tid; idx < 64 * 32; idx += NTHREADS) {
      const int k = idx >> 6, d = idx & 63;
      Bs[d * LDT + k] = QKV[(rowb + k) * 1536 + 1024 + h * 64 + d];  // v slice
    }
    __syncthreads();
    if (tid < 64) {                          // fold ksum out of the staged tile
#pragma unroll
      for (int k = 0; k < 32; ++k) ksacc += (float)As[tid * LDT + k];
    }
    const v16bf af = load_frag_a(As, LDT, wave * 16, lane);
#pragma unroll
    for (int g = 0; g < 4; ++g) {
      const v16bf bv = load_frag_b(Bs, LDT, g * 16, lane);
      acc[g] = wmma_bf16f32(af, bv, acc[g]);
    }
    __syncthreads();
  }
  const int hh = lane >> 4, nc = lane & 15;
#pragma unroll
  for (int g = 0; g < 4; ++g) {
#pragma unroll
    for (int i = 0; i < 8; ++i) {
      const int f = wave * 16 + i + 8 * hh;
      atomicAdd(&kvbuf[(bh * 64 + f) * 64 + g * 16 + nc], acc[g][i]);
    }
  }
  if (tid < 64) atomicAdd(&ksum[bh * 64 + tid], ksacc);
}

// ---------------- stage 4: attn = (qphi @ kv) * z -------------------------------
__global__ __launch_bounds__(NTHREADS) void k_attn(const bf16* __restrict__ qphi,
                                                   const float* __restrict__ kvbuf,
                                                   const float* __restrict__ ksum,
                                                   bf16* __restrict__ attn) {
  __shared__ __align__(16) bf16 As[64 * LDT2];  // [row][f]
  __shared__ __align__(16) bf16 Bs[64 * LDT2];  // [d][f]
  __shared__ float ksm[64];
  __shared__ float zz[64];
  const int tid = threadIdx.x, lane = tid & 31, wave = tid >> 5;
  const long m0 = (long)blockIdx.x * 64;
  const int h = blockIdx.y;
  const int bh = (int)(m0 >> 12) * 8 + h;
  const bf16* src = qphi + m0 * 512 + h * 64;
  for (int idx = tid; idx < 64 * 32; idx += NTHREADS) {
    const int r = idx >> 5, fp = (idx & 31) << 1;
    uint32_t v; __builtin_memcpy(&v, src + (long)r * 512 + fp, 4);
    __builtin_memcpy(&As[r * LDT2 + fp], &v, 4);
  }
  for (int idx = tid; idx < 64 * 64; idx += NTHREADS) {     // B^T: Bs[d][f] = kv[f][d]
    const int f = idx >> 6, d = idx & 63;
    Bs[d * LDT2 + f] = (bf16)kvbuf[(bh * 64 + f) * 64 + d];
  }
  if (tid < 64) ksm[tid] = ksum[bh * 64 + tid];
  __syncthreads();
  if (tid < 64) {                                           // z = 1/(qphi . ksum)
    float s = 0.f;
#pragma unroll
    for (int f = 0; f < 64; ++f) s += (float)As[tid * LDT2 + f] * ksm[f];
    zz[tid] = 1.0f / s;
  }
  v8f acc[4] = {};
#pragma unroll
  for (int ks = 0; ks < 64; ks += 32) {
    const v16bf af = load_frag_a(As + ks, LDT2, wave * 16, lane);
#pragma unroll
    for (int g = 0; g < 4; ++g) {
      const v16bf bv = load_frag_b(Bs + ks, LDT2, g * 16, lane);
      acc[g] = wmma_bf16f32(af, bv, acc[g]);
    }
  }
  __syncthreads();                                          // zz visible to all
  const int hh = lane >> 4, nc = lane & 15;
#pragma unroll
  for (int g = 0; g < 4; ++g) {
#pragma unroll
    for (int i = 0; i < 8; ++i) {
      const int rl = wave * 16 + i + 8 * hh;
      attn[(m0 + rl) * 512 + h * 64 + g * 16 + nc] = (bf16)(acc[g][i] * zz[rl]);
    }
  }
}

// ---------------- stage 5: out = attn @ w_proj + b_proj (f32 out) ---------------
__global__ __launch_bounds__(NTHREADS) void k_gemm_out(const bf16* __restrict__ A,
                                                       const bf16* __restrict__ W,
                                                       const float* __restrict__ bias,
                                                       float* __restrict__ Out) {
  __shared__ __align__(16) bf16 As[128 * LDT];
  __shared__ __align__(16) bf16 Bs[64 * LDT];
  const int tid = threadIdx.x, lane = tid & 31, wave = tid >> 5;
  const long m0 = (long)blockIdx.x * 128;
  const int  n0 = blockIdx.y * 64;
  v8f acc[2][4] = {};
  for (int k0 = 0; k0 < 512; k0 += 32) {
#if USE_TDM
    if (wave == 0) tdm_load_tile(A + m0 * 512 + k0, (uint32_t)(uintptr_t)As, 128);
#else
    for (int idx = tid; idx < 128 * 16; idx += NTHREADS) {
      const int r = idx >> 4, kp = (idx & 15) << 1;
      uint32_t v; __builtin_memcpy(&v, A + (m0 + r) * 512 + k0 + kp, 4);
      __builtin_memcpy(&As[r * LDT + kp], &v, 4);
    }
#endif
    for (int idx = tid; idx < 32 * 32; idx += NTHREADS) {
      const int k = idx >> 5, n2 = (idx & 31) << 1;
      uint32_t v; __builtin_memcpy(&v, W + (long)(k0 + k) * 512 + n0 + n2, 4);
      bf16 two[2]; __builtin_memcpy(two, &v, 4);
      Bs[n2 * LDT + k] = two[0];
      Bs[(n2 + 1) * LDT + k] = two[1];
    }
#if USE_TDM
    if (wave == 0) __builtin_amdgcn_s_wait_tensorcnt(0);
#endif
    __syncthreads();
    const v16bf a0 = load_frag_a(As, LDT, wave * 32, lane);
    const v16bf a1 = load_frag_a(As, LDT, wave * 32 + 16, lane);
#pragma unroll
    for (int g = 0; g < 4; ++g) {
      const v16bf bv = load_frag_b(Bs, LDT, g * 16, lane);
      acc[0][g] = wmma_bf16f32(a0, bv, acc[0][g]);
      acc[1][g] = wmma_bf16f32(a1, bv, acc[1][g]);
    }
    __syncthreads();
  }
  const int hh = lane >> 4, nc = lane & 15;
#pragma unroll
  for (int s = 0; s < 2; ++s) {
#pragma unroll
    for (int g = 0; g < 4; ++g) {
#pragma unroll
      for (int i = 0; i < 8; ++i) {
        const long row = m0 + wave * 32 + s * 16 + i + 8 * hh;
        const int col = n0 + g * 16 + nc;
        Out[row * 512 + col] = acc[s][g][i] + bias[col];
      }
    }
  }
}

extern "C" void kernel_launch(void* const* d_in, const int* in_sizes, int n_in,
                              void* d_out, int out_size, void* d_ws, size_t ws_size,
                              hipStream_t stream) {
  (void)in_sizes; (void)n_in; (void)out_size; (void)ws_size;
  const float* x      = (const float*)d_in[0];
  const float* w_qkv  = (const float*)d_in[1];
  const float* proj   = (const float*)d_in[2];
  const float* w_proj = (const float*)d_in[3];
  const float* b_proj = (const float*)d_in[4];
  float* out = (float*)d_out;

  const long M = 8L * 4096;  // B*N rows
  char* base = (char*)d_ws;
  size_t off = 0;
  auto take = [&](size_t bytes) -> char* {
    char* p = base + off;
    off = (off + bytes + 255) & ~(size_t)255;
    return p;
  };
  bf16* x_bf     = (bf16*)take(M * 512 * 2);        // 32 MB
  bf16* wqkv_bf  = (bf16*)take(512L * 1536 * 2);
  bf16* proj_bf  = (bf16*)take(8L * 64 * 64 * 2);
  bf16* wproj_bf = (bf16*)take(512L * 512 * 2);
  bf16* qkv_bf   = (bf16*)take(M * 1536 * 2);       // 96 MB
  bf16* qphi     = (bf16*)take(M * 512 * 2);        // 32 MB
  bf16* kphi     = (bf16*)take(M * 512 * 2);        // 32 MB
  float* kvbuf   = (float*)take(64L * 64 * 64 * 4); // [b*h][f][d]
  float* ksum    = (float*)take(64L * 64 * 4);
  bf16* attn     = (bf16*)take(M * 512 * 2);        // 32 MB

  k_cvt<<<dim3((unsigned)((M * 512 + 255) / 256)), 256, 0, stream>>>(x, x_bf, (int)(M * 512));
  k_cvt<<<dim3((512 * 1536 + 255) / 256), 256, 0, stream>>>(w_qkv, wqkv_bf, 512 * 1536);
  k_cvt<<<dim3((8 * 64 * 64 + 255) / 256), 256, 0, stream>>>(proj, proj_bf, 8 * 64 * 64);
  k_cvt<<<dim3((512 * 512 + 255) / 256), 256, 0, stream>>>(w_proj, wproj_bf, 512 * 512);
  k_zero<<<dim3((64 * 64 * 64 + 255) / 256), 256, 0, stream>>>(kvbuf, 64 * 64 * 64);
  k_zero<<<dim3((64 * 64 + 255) / 256), 256, 0, stream>>>(ksum, 64 * 64);

  k_gemm_qkv<<<dim3(256, 24), NTHREADS, 0, stream>>>(x_bf, wqkv_bf, qkv_bf);
  k_phi<<<dim3(512, 8, 2), NTHREADS, 0, stream>>>(qkv_bf, proj_bf, qphi, kphi);
  k_kv<<<dim3(64, 8), NTHREADS, 0, stream>>>(kphi, qkv_bf, kvbuf, ksum);
  k_attn<<<dim3(512, 8), NTHREADS, 0, stream>>>(qphi, kvbuf, ksum, attn);
  k_gemm_out<<<dim3(256, 8), NTHREADS, 0, stream>>>(attn, wproj_bf, b_proj, out);
}
